// MultiTaskGATv2_6210522710527
// MI455X (gfx1250) — compile-verified, matrix operand order checked by reading, not run on hardware
//
#include <hip/hip_runtime.h>
#include <hip/hip_bf16.h>

// ---------- sizes ----------
#define NN      20000
#define EE      320000
#define ETOT    (EE + NN)     // edges + self loops
#define GG      8
#define FINF    16
#define HID     256
#define HEADS   8
#define CPH     32
#define LAYERS  6
#define MAXC    200
#define NTYPES  8
#define LNEPS   1e-5f

// ---------- CDNA5 WMMA types ----------
typedef __attribute__((ext_vector_type(16))) __bf16 v16bf;
typedef __attribute__((ext_vector_type(8)))  float  v8f;
typedef __attribute__((ext_vector_type(4)))  unsigned int u32x4;
typedef __attribute__((ext_vector_type(4)))  int i32x4;

union Frag { v16bf v; u32x4 q[2]; };

// ---------- gfx1250 async-to-LDS path (guarded; falls back to ds_store staging) ----
#if defined(__HIP_DEVICE_COMPILE__) && \
    __has_builtin(__builtin_amdgcn_global_load_async_to_lds_b128) && \
    __has_builtin(__builtin_amdgcn_s_wait_asynccnt)
#define USE_ASYNC_LDS 1
#else
#define USE_ASYNC_LDS 0
#endif

__device__ __forceinline__ void async_copy16(const unsigned short* g, unsigned short* l) {
#if USE_ASYNC_LDS
  // signature (from hipcc diagnostic): (global v4i*, local v4i*, imm offset, imm cpol)
  __builtin_amdgcn_global_load_async_to_lds_b128(
      (__attribute__((address_space(1))) i32x4*)g,
      (__attribute__((address_space(3))) i32x4*)l, 0, 0);
#else
  *(u32x4*)l = *(const u32x4*)g;
#endif
}

__device__ __forceinline__ void async_wait_all() {
#if USE_ASYNC_LDS
  __builtin_amdgcn_s_wait_asynccnt(0);
#endif
}

// ---------- helpers ----------
__device__ __forceinline__ unsigned short f2bf(float f) {
  unsigned int u = __float_as_uint(f);
  u += 0x7FFFu + ((u >> 16) & 1u);           // round-to-nearest-even
  return (unsigned short)(u >> 16);
}

__device__ __forceinline__ void atomicMaxF(float* addr, float val) {
  // ordered-int trick; works for +/- values with -inf init
  if (val >= 0.0f) atomicMax((int*)addr, __float_as_int(val));
  else             atomicMin((unsigned int*)addr, __float_as_uint(val));
}

// ---------- generic fill ----------
__global__ void k_fill(float* p, float v, int n) {
  int i = blockIdx.x * blockDim.x + threadIdx.x;
  if (i < n) p[i] = v;
}

// ---------- f32 -> bf16 convert ----------
__global__ void k_f2bf(const float* __restrict__ in, unsigned short* __restrict__ out, int n) {
  int i = blockIdx.x * blockDim.x + threadIdx.x;
  if (i < n) out[i] = f2bf(in[i]);
}

// ---------- batched transpose + convert: out[b][n][k] = bf16(W[b][k][n]) ----------
__global__ void k_tconv(const float* __restrict__ W, unsigned short* __restrict__ out,
                        int total, int K, int Nc) {
  int i = blockIdx.x * blockDim.x + threadIdx.x;
  if (i >= total) return;
  int b = i / (K * Nc);
  int r = i - b * K * Nc;
  int n = r / K;
  int k = r - n * K;
  out[i] = f2bf(W[(size_t)b * K * Nc + (size_t)k * Nc + n]);
}

// ---------- input projection: h = x @ in_w + in_b (K=16, scalar is fine) ----------
__global__ void k_inproj(const float* __restrict__ x, const float* __restrict__ w,
                         const float* __restrict__ b, float* __restrict__ h) {
  int i = blockIdx.x * blockDim.x + threadIdx.x;
  if (i >= NN * HID) return;
  int n = i >> 8, j = i & 255;
  float acc = b[j];
  #pragma unroll
  for (int k = 0; k < FINF; k++) acc += x[n * FINF + k] * w[k * HID + j];
  h[i] = acc;
}

// ---------- WMMA GEMM: C[M,Nc] = A(bf16)[M,K] @ Bt(bf16)[Nc,K]^T + bias, opt relu ----------
// block = 128 threads = 4 waves; wave => 16x64 tile (4 accumulators).
// B tile (64 cols x K) staged once per block in LDS (async-to-LDS when available);
// fragments are then fed from LDS (DScnt) while A streams from global (LOADcnt).
template<int KDIM>
__global__ __launch_bounds__(128)
void k_wmma_gemm(const unsigned short* __restrict__ A, const unsigned short* __restrict__ Bt,
                 const float* __restrict__ bias, float* __restrict__ Cout,
                 int M, int Nc, int doRelu) {
  constexpr int BPAD = 8;               // 16B pad -> fragment lanes hit distinct banks
  constexpr int BSTR = KDIM + BPAD;     // LDS row stride in elements
  __shared__ unsigned short Bs[64 * BSTR];

  const int tid  = threadIdx.x;
  const int wave = tid >> 5;
  const int lane = tid & 31;
  const int half = lane >> 4;           // which K-half this lane holds
  const int lm   = lane & 15;
  const int row0 = blockIdx.x * 64 + wave * 16;
  const int col0 = blockIdx.y * 64;

  // ---- stage B tile: 64 x KDIM bf16 = 2048 chunks of 16B, 16 per thread ----
  constexpr int CHUNKS = 64 * KDIM / 8;
  #pragma unroll 4
  for (int c = tid; c < CHUNKS; c += 128) {
    int bcolL = c / (KDIM / 8);
    int kblk  = c - bcolL * (KDIM / 8);
    int gcol  = col0 + bcolL; if (gcol >= Nc) gcol = Nc - 1;   // clamp, store-side guarded
    async_copy16(Bt + (size_t)gcol * KDIM + kblk * 8,
                 Bs + bcolL * BSTR + kblk * 8);
  }
  async_wait_all();
  __syncthreads();

  int arow = row0 + lm; if (arow >= M) arow = M - 1;           // clamp (no EXEC change)
  const unsigned short* aptr = A + (size_t)arow * KDIM + half * 8;

  v8f acc[4] = {};
  #pragma unroll
  for (int k0 = 0; k0 < KDIM; k0 += 32) {
    Frag af;
    af.q[0] = *(const u32x4*)(aptr + k0);
    af.q[1] = *(const u32x4*)(aptr + k0 + 16);
    if (k0 + 64 < KDIM) __builtin_prefetch(aptr + k0 + 64, 0, 1);  // global_prefetch_b8
    #pragma unroll
    for (int j = 0; j < 4; j++) {
      const unsigned short* bp = Bs + (j * 16 + lm) * BSTR + half * 8 + k0;
      Frag bfg;
      bfg.q[0] = *(const u32x4*)(bp);
      bfg.q[1] = *(const u32x4*)(bp + 16);
      acc[j] = __builtin_amdgcn_wmma_f32_16x16x32_bf16(
                 false, af.v, false, bfg.v, (short)0, acc[j], false, false);
    }
  }

  // C/D layout: VGPR i -> M = i (lanes 0-15) / i+8 (lanes 16-31), N = lane&15
  #pragma unroll
  for (int j = 0; j < 4; j++) {
    int c = col0 + j * 16 + lm;
    if (c < Nc) {
      float bv = bias ? bias[c] : 0.0f;
      #pragma unroll
      for (int i = 0; i < 8; i++) {
        int r = row0 + half * 8 + i;
        if (r < M) {
          float v = acc[j][i] + bv;
          if (doRelu) v = fmaxf(v, 0.0f);
          Cout[(size_t)r * Nc + c] = v;
        }
      }
    }
  }
}

// ---------- GAT phase 1: logits + segment max ----------
__global__ void k_gat_logits(const float* __restrict__ xl, const float* __restrict__ xr,
                             const int* __restrict__ ei, const float* __restrict__ att,
                             float* __restrict__ attb, float* __restrict__ mmax) {
  int tid = blockIdx.x * blockDim.x + threadIdx.x;
  if (tid >= ETOT * HEADS) return;
  int e = tid >> 3, hd = tid & 7;
  int src, dst;
  if (e < EE) { src = ei[e]; dst = ei[EE + e]; } else { src = e - EE; dst = src; }
  const float* pl = xl + (size_t)src * HID + hd * CPH;
  const float* pr = xr + (size_t)dst * HID + hd * CPH;
  const float* pa = att + hd * CPH;
  float lg = 0.0f;
  #pragma unroll
  for (int c = 0; c < CPH; c++) {
    float t = pl[c] + pr[c];
    t = (t > 0.0f) ? t : 0.2f * t;      // leaky_relu(0.2)
    lg += t * pa[c];
  }
  attb[tid] = lg;
  atomicMaxF(&mmax[dst * HEADS + hd], lg);
}

// ---------- GAT phase 2: exp + segment sum ----------
__global__ void k_gat_expsum(const int* __restrict__ ei, float* __restrict__ attb,
                             const float* __restrict__ mmax, float* __restrict__ ssum) {
  int tid = blockIdx.x * blockDim.x + threadIdx.x;
  if (tid >= ETOT * HEADS) return;
  int e = tid >> 3, hd = tid & 7;
  int dst = (e < EE) ? ei[EE + e] : (e - EE);
  float a = __expf(attb[tid] - mmax[dst * HEADS + hd]);
  attb[tid] = a;
  atomicAdd(&ssum[dst * HEADS + hd], a);
}

// ---------- GAT phase 3: alpha-weighted aggregation ----------
__global__ void k_gat_agg(const float* __restrict__ xl, const int* __restrict__ ei,
                          const float* __restrict__ attb, const float* __restrict__ ssum,
                          float* __restrict__ gout) {
  int tid = blockIdx.x * blockDim.x + threadIdx.x;
  if (tid >= ETOT * HID) return;
  int e = tid >> 8, cc = tid & 255, hd = cc >> 5;
  int src, dst;
  if (e < EE) { src = ei[e]; dst = ei[EE + e]; } else { src = e - EE; dst = src; }
  float alpha = attb[e * HEADS + hd] / ssum[dst * HEADS + hd];
  atomicAdd(&gout[(size_t)dst * HID + cc], xl[(size_t)src * HID + cc] * alpha);
}

// ---------- GAT phase 4: bias + ELU + LayerNorm + residual (wave32 per node) ----------
__global__ __launch_bounds__(256)
void k_gat_norm(const float* __restrict__ gout, const float* __restrict__ gb,
                const float* __restrict__ lg, const float* __restrict__ lb,
                float* __restrict__ h) {
  int wave = threadIdx.x >> 5, lane = threadIdx.x & 31;
  int n = blockIdx.x * 8 + wave;                 // NN divisible by 8
  float v[8]; float s = 0.0f;
  #pragma unroll
  for (int t = 0; t < 8; t++) {
    int c = t * 32 + lane;
    float x = gout[(size_t)n * HID + c] + gb[c];
    x = (x > 0.0f) ? x : expm1f(x);              // elu
    v[t] = x; s += x;
  }
  #pragma unroll
  for (int o = 16; o > 0; o >>= 1) s += __shfl_xor(s, o, 32);
  float mu = s * (1.0f / HID);
  float var = 0.0f;
  #pragma unroll
  for (int t = 0; t < 8; t++) { float d = v[t] - mu; var += d * d; }
  #pragma unroll
  for (int o = 16; o > 0; o >>= 1) var += __shfl_xor(var, o, 32);
  var *= (1.0f / HID);
  float inv = rsqrtf(var + LNEPS);
  #pragma unroll
  for (int t = 0; t < 8; t++) {
    int c = t * 32 + lane;
    float hn = (v[t] - mu) * inv * lg[c] + lb[c];
    h[(size_t)n * HID + c] += hn;                // residual
  }
}

// ---------- graph pooling ----------
__global__ void k_pool(const float* __restrict__ h, const int* __restrict__ batch,
                       float* __restrict__ gsum, float* __restrict__ gmax,
                       float* __restrict__ cnt) {
  int i = blockIdx.x * blockDim.x + threadIdx.x;
  if (i >= NN * HID) return;
  int n = i >> 8, c = i & 255;
  int b = batch[n];
  float v = h[i];
  atomicAdd(&gsum[b * HID + c], v);
  atomicMaxF(&gmax[b * HID + c], v);
  if (c == 0) atomicAdd(&cnt[b], 1.0f);
}

__global__ void k_grep(const float* __restrict__ gsum, const float* __restrict__ gmax,
                       const float* __restrict__ cnt, float* __restrict__ grep) {
  int i = blockIdx.x * blockDim.x + threadIdx.x;
  if (i >= GG * 2 * HID) return;
  int g = i >> 9, c = i & 511;
  grep[i] = (c < HID) ? gsum[g * HID + c] / fmaxf(cnt[g], 1.0f)
                      : gmax[g * HID + (c - HID)];
}

// ---------- small scalar GEMM for the heads (M=8) ----------
__global__ void k_sgemm(const float* __restrict__ A, const float* __restrict__ W,
                        const float* __restrict__ b, float* __restrict__ Out,
                        int M, int K, int Nc, int doRelu) {
  int i = blockIdx.x * blockDim.x + threadIdx.x;
  if (i >= M * Nc) return;
  int m = i / Nc, j = i - m * Nc;
  float acc = b[j];
  for (int k = 0; k < K; k++) acc += A[m * K + k] * W[k * Nc + j];
  if (doRelu) acc = fmaxf(acc, 0.0f);
  Out[i] = acc;
}

// ---------- host orchestration ----------
extern "C" void kernel_launch(void* const* d_in, const int* in_sizes, int n_in,
                              void* d_out, int out_size, void* d_ws, size_t ws_size,
                              hipStream_t stream) {
  const float* x     = (const float*)d_in[0];
  const int*   ei    = (const int*)  d_in[1];
  const int*   batch = (const int*)  d_in[2];
  const float* in_w  = (const float*)d_in[3];
  const float* in_b  = (const float*)d_in[4];
  const float* Wl    = (const float*)d_in[5];
  const float* bl    = (const float*)d_in[6];
  const float* Wr    = (const float*)d_in[7];
  const float* br    = (const float*)d_in[8];
  const float* att   = (const float*)d_in[9];
  const float* gat_b = (const float*)d_in[10];
  const float* ln_g  = (const float*)d_in[11];
  const float* ln_b  = (const float*)d_in[12];
  const float* cw1 = (const float*)d_in[13]; const float* cb1 = (const float*)d_in[14];
  const float* cw2 = (const float*)d_in[15]; const float* cb2 = (const float*)d_in[16];
  const float* qw1 = (const float*)d_in[17]; const float* qb1 = (const float*)d_in[18];
  const float* qw2 = (const float*)d_in[19]; const float* qb2 = (const float*)d_in[20];
  const float* qw3 = (const float*)d_in[21]; const float* qb3 = (const float*)d_in[22];
  const float* tw1 = (const float*)d_in[23]; const float* tb1 = (const float*)d_in[24];
  const float* tw2 = (const float*)d_in[25]; const float* tb2 = (const float*)d_in[26];
  const float* tw3 = (const float*)d_in[27]; const float* tb3 = (const float*)d_in[28];
  const float* dw1 = (const float*)d_in[29]; const float* db1 = (const float*)d_in[30];
  const float* dw2 = (const float*)d_in[31]; const float* db2 = (const float*)d_in[32];
  const float* dw3 = (const float*)d_in[33]; const float* db3 = (const float*)d_in[34];

  char* ws = (char*)d_ws;
  size_t off = 0;
  auto take = [&](size_t bytes) -> void* {
    void* p = ws + off;
    off = (off + bytes + 255) & ~(size_t)255;
    return p;
  };
  float*          h    = (float*)take((size_t)NN * HID * 4);
  unsigned short* abf  = (unsigned short*)take((size_t)NN * HID * 2);
  float*          xl   = (float*)take((size_t)NN * HID * 4);
  float*          xr   = (float*)take((size_t)NN * HID * 4);
  float*          attb = (float*)take((size_t)ETOT * HEADS * 4);
  float*          mmax = (float*)take((size_t)NN * HEADS * 4);
  float*          ssum = (float*)take((size_t)NN * HEADS * 4);
  float*          gout = (float*)take((size_t)NN * HID * 4);
  unsigned short* WlT  = (unsigned short*)take((size_t)LAYERS * HID * HID * 2);
  unsigned short* WrT  = (unsigned short*)take((size_t)LAYERS * HID * HID * 2);
  unsigned short* cw1T = (unsigned short*)take((size_t)HID * HID * 2);
  unsigned short* cw2T = (unsigned short*)take((size_t)HID * MAXC * 2);
  unsigned short* hbf2 = (unsigned short*)take((size_t)NN * HID * 2);
  float*          gsum = (float*)take((size_t)GG * HID * 4);
  float*          gmax = (float*)take((size_t)GG * HID * 4);
  float*          cnt  = (float*)take((size_t)GG * 4);
  float*          grep = (float*)take((size_t)GG * 2 * HID * 4);
  float*          t1   = (float*)take((size_t)GG * HID * 4);
  float*          t2   = (float*)take((size_t)GG * 64 * 4);

  auto cdiv = [](int a, int b) { return (a + b - 1) / b; };

  // weight prep (bf16, transposed so B frags load contiguously along K)
  k_tconv<<<cdiv(LAYERS*HID*HID,256),256,0,stream>>>(Wl,  WlT,  LAYERS*HID*HID, HID, HID);
  k_tconv<<<cdiv(LAYERS*HID*HID,256),256,0,stream>>>(Wr,  WrT,  LAYERS*HID*HID, HID, HID);
  k_tconv<<<cdiv(HID*HID,256),256,0,stream>>>(cw1, cw1T, HID*HID,  HID, HID);
  k_tconv<<<cdiv(HID*MAXC,256),256,0,stream>>>(cw2, cw2T, MAXC*HID, HID, MAXC);

  // input projection
  k_inproj<<<cdiv(NN*HID,256),256,0,stream>>>(x, in_w, in_b, h);

  dim3 gdim(cdiv(NN,64), cdiv(HID,64));
  for (int l = 0; l < LAYERS; l++) {
    k_f2bf<<<cdiv(NN*HID,256),256,0,stream>>>(h, abf, NN*HID);
    k_wmma_gemm<HID><<<gdim,128,0,stream>>>(abf, WlT + (size_t)l*HID*HID, bl + l*HID, xl, NN, HID, 0);
    k_wmma_gemm<HID><<<gdim,128,0,stream>>>(abf, WrT + (size_t)l*HID*HID, br + l*HID, xr, NN, HID, 0);
    k_fill<<<cdiv(NN*HEADS,256),256,0,stream>>>(mmax, -INFINITY, NN*HEADS);
    k_fill<<<cdiv(NN*HEADS,256),256,0,stream>>>(ssum, 0.0f, NN*HEADS);
    k_fill<<<cdiv(NN*HID,256),256,0,stream>>>(gout, 0.0f, NN*HID);
    k_gat_logits<<<cdiv(ETOT*HEADS,256),256,0,stream>>>(xl, xr, ei, att + l*HEADS*CPH, attb, mmax);
    k_gat_expsum<<<cdiv(ETOT*HEADS,256),256,0,stream>>>(ei, attb, mmax, ssum);
    k_gat_agg<<<cdiv(ETOT*HID,256),256,0,stream>>>(xl, ei, attb, ssum, gout);
    k_gat_norm<<<NN/8,256,0,stream>>>(gout, gat_b + l*HID, ln_g + l*HID, ln_b + l*HID, h);
  }

  // color head: relu(h@cw1+cb1) @ cw2 + cb2 -> d_out[0 .. N*MAXC)
  float* out_color = (float*)d_out;
  k_f2bf<<<cdiv(NN*HID,256),256,0,stream>>>(h, abf, NN*HID);
  k_wmma_gemm<HID><<<gdim,128,0,stream>>>(abf, cw1T, cb1, xl, NN, HID, 1);
  k_f2bf<<<cdiv(NN*HID,256),256,0,stream>>>(xl, hbf2, NN*HID);
  dim3 gdim2(cdiv(NN,64), cdiv(MAXC,64));
  k_wmma_gemm<HID><<<gdim2,128,0,stream>>>(hbf2, cw2T, cb2, out_color, NN, MAXC, 0);

  // pooling
  k_fill<<<cdiv(GG*HID,256),256,0,stream>>>(gsum, 0.0f, GG*HID);
  k_fill<<<cdiv(GG*HID,256),256,0,stream>>>(gmax, -INFINITY, GG*HID);
  k_fill<<<1,256,0,stream>>>(cnt, 0.0f, GG);
  k_pool<<<cdiv(NN*HID,256),256,0,stream>>>(h, batch, gsum, gmax, cnt);
  k_grep<<<cdiv(GG*2*HID,256),256,0,stream>>>(gsum, gmax, cnt, grep);

  // heads
  float* out_chrom = out_color + (size_t)NN * MAXC;
  float* out_type  = out_chrom + GG;
  float* out_diff  = out_type  + GG * NTYPES;

  k_sgemm<<<cdiv(GG*HID,256),256,0,stream>>>(grep, qw1, qb1, t1, GG, 2*HID, HID, 1);
  k_sgemm<<<cdiv(GG*64,256),256,0,stream>>>(t1, qw2, qb2, t2, GG, HID, 64, 1);
  k_sgemm<<<1,256,0,stream>>>(t2, qw3, qb3, out_chrom, GG, 64, 1, 0);

  k_sgemm<<<cdiv(GG*HID,256),256,0,stream>>>(grep, tw1, tb1, t1, GG, 2*HID, HID, 1);
  k_sgemm<<<cdiv(GG*64,256),256,0,stream>>>(t1, tw2, tb2, t2, GG, HID, 64, 1);
  k_sgemm<<<1,256,0,stream>>>(t2, tw3, tb3, out_type, GG, 64, NTYPES, 0);

  k_sgemm<<<cdiv(GG*HID,256),256,0,stream>>>(grep, dw1, db1, t1, GG, 2*HID, HID, 1);
  k_sgemm<<<cdiv(GG*64,256),256,0,stream>>>(t1, dw2, db2, t2, GG, HID, 64, 1);
  k_sgemm<<<1,256,0,stream>>>(t2, dw3, db3, out_diff, GG, 64, 1, 0);

  (void)in_sizes; (void)n_in; (void)out_size; (void)ws_size;
}
